// T2VTTF_87754771792010
// MI455X (gfx1250) — matrix operand order repaired and checked by reading, not compiled, not run on hardware
//
#include <hip/hip_runtime.h>
#include <hip/hip_bf16.h>

#define DEVINL static __device__ __forceinline__

constexpr int Bx  = 32;
constexpr int Nx  = 2048;
constexpr int Dx  = 768;
constexpr int Fx  = 1152;   // D + D_TAU
constexpr int Hx  = 8;
constexpr int HDx = 96;

typedef __bf16 bf16x16 __attribute__((ext_vector_type(16)));
typedef float  f32x8   __attribute__((ext_vector_type(8)));
typedef unsigned int u32x4 __attribute__((ext_vector_type(4)));
typedef int          i32x4 __attribute__((ext_vector_type(4)));
typedef int          i32x8 __attribute__((ext_vector_type(8)));

union BFrag { bf16x16 v; unsigned int u[8]; };
union FAcc  { f32x8 v;   float f[8]; };

DEVINL unsigned short f2bf(float f) {
  unsigned int u = __float_as_uint(f);
  u += 0x7FFFu + ((u >> 16) & 1u);           // round-to-nearest-even
  return (unsigned short)(u >> 16);
}

DEVINL f32x8 zero8() {
  f32x8 z = {0.f, 0.f, 0.f, 0.f, 0.f, 0.f, 0.f, 0.f};
  return z;
}

DEVINL f32x8 wmma_bf16(const BFrag& a, const BFrag& b, f32x8 c) {
  // (neg_a, A, neg_b, B, c_mod, C, reuse_a, reuse_b)
  return __builtin_amdgcn_wmma_f32_16x16x32_bf16(false, a.v, false, b.v,
                                                 (short)0, c, false, false);
}

// ---------------------------------------------------------------------------
// Tensor Data Mover: 2D tile (tile0 elements per row, tile1 rows, row stride
// stride0 elements) from global -> LDS. D# packed per CDNA5 ISA:
//   g0: [1:0]=count=1, [63:32]=lds_addr, [120:64]=global_addr, [127:126]=2
//   g1: [17:16]=data_size, [79:48]=tensor_dim0, [111:80]=tensor_dim1,
//       [127:112]=tile_dim0, [143:128]=tile_dim1, [207:160]=tensor_dim0_stride
// Issued by one wave; completion via TENSORcnt (one DMA in flight -> wait 0).
// clang-23 lane uses the 6-arg builtin:
//   (u32x4 g0, i32x8 g1, i32x4 g2, i32x4 g3, i32x8 extra, i32 cpol)
// ---------------------------------------------------------------------------
DEVINL void tdm_load_2d(unsigned lds_addr, const void* gptr, int data_size_code,
                        int tile0, int tile1, long long stride0) {
  const unsigned long long ga = (unsigned long long)gptr;
  u32x4 g0;
  g0.x = 1u;                                            // count=1, user mode
  g0.y = lds_addr;                                      // LDS byte address
  g0.z = (unsigned)ga;                                  // global_addr[31:0]
  g0.w = (unsigned)((ga >> 32) & 0x01FFFFFFu) | (2u << 30);  // [56:32] | type=2
  i32x8 g1;
  g1[0] = data_size_code << 16;                         // mask=0, data_size
  g1[1] = (int)(((unsigned)tile0 & 0xFFFFu) << 16);     // tensor_dim0[15:0]
  g1[2] = (int)((((unsigned)tile0 >> 16) & 0xFFFFu) |
                (((unsigned)tile1 & 0xFFFFu) << 16));   // tdim0 hi | tdim1 lo
  g1[3] = (int)((((unsigned)tile1 >> 16) & 0xFFFFu) |
                (((unsigned)tile0 & 0xFFFFu) << 16));   // tdim1 hi | tile_dim0
  g1[4] = (int)((unsigned)tile1 & 0xFFFFu);             // tile_dim1, tile_dim2=0
  g1[5] = (int)(stride0 & 0xFFFFFFFFll);                // tensor_dim0_stride lo
  g1[6] = (int)((stride0 >> 32) & 0xFFFFll);            // tensor_dim0_stride hi
  g1[7] = 0;
  const i32x4 z4 = {0, 0, 0, 0};
  const i32x8 z8 = {0, 0, 0, 0, 0, 0, 0, 0};
  __builtin_amdgcn_tensor_load_to_lds(g0, g1, z4, z4, z8, 0);
}

// ---------------------------------------------------------------------------
// Kernel 0a: q = Qp @ Wq^T + bq ; qk_w[h,f] = q_h . Wk_h[:,f] ; qb[h] = q_h.bk_h
// ---------------------------------------------------------------------------
__global__ __launch_bounds__(256)
void prep_q_kernel(const float* __restrict__ Qp, const float* __restrict__ in_w,
                   const float* __restrict__ in_b, float* __restrict__ qv,
                   float* __restrict__ qkw, float* __restrict__ qb) {
  __shared__ float sQp[Dx];
  __shared__ float sq[Dx];
  const int tid = threadIdx.x;
  for (int d = tid; d < Dx; d += 256) sQp[d] = Qp[d];
  __syncthreads();
  for (int d = tid; d < Dx; d += 256) {
    const float* w = in_w + (size_t)d * Dx;
    float a = in_b[d];
    for (int f = 0; f < Dx; ++f) a += sQp[f] * w[f];
    sq[d] = a;
    qv[d] = a;
  }
  __syncthreads();
  for (int o = tid; o < Hx * Dx; o += 256) {
    const int h = o / Dx, f = o - h * Dx;
    float a = 0.f;
    for (int d = 0; d < HDx; ++d)
      a += sq[h * HDx + d] * in_w[(size_t)(Dx + h * HDx + d) * Dx + f];
    qkw[o] = a;
  }
  if (tid < Hx) {
    float a = 0.f;
    for (int d = 0; d < HDx; ++d)
      a += sq[tid * HDx + d] * in_b[Dx + tid * HDx + d];
    qb[tid] = a;
  }
}

// ---------------------------------------------------------------------------
// Kernel 0b: f32 -> bf16 bulk convert (kv_W)
// ---------------------------------------------------------------------------
__global__ __launch_bounds__(256)
void cvt_bf16_kernel(const float* __restrict__ src,
                     unsigned short* __restrict__ dst, int n) {
  const int i = blockIdx.x * 256 + threadIdx.x;
  if (i < n) dst[i] = f2bf(src[i]);
}

// ---------------------------------------------------------------------------
// Kernel 1: fused time2vec + KV GEMM (bf16 WMMA) + score epilogue.
// Block = 256 thr (8 waves), tile = 32 rows x 768 cols, K = 1152 in 32-chunks.
// Wave w: m-tile (w&1), col group (w>>1)*192, 12 n-tiles of 16.
// kv_W K-slabs double-buffered through LDS by the Tensor Data Mover: wave 0
// waits TENSORcnt for the current buffer, the workgroup barrier publishes it,
// then the next chunk's DMA is issued into the other buffer while all waves
// compute. KV stored bf16 chunk-transposed [b][n/32][d][n%32] (NT stores).
// ---------------------------------------------------------------------------
__global__ __launch_bounds__(256)
void kv_gemm_scores_kernel(const float* __restrict__ text_embs,
                           const float* __restrict__ text_ts,
                           const float* __restrict__ t2v_w0,
                           const float* __restrict__ t2v_b0,
                           const float* __restrict__ t2v_Wp,
                           const float* __restrict__ t2v_bp,
                           const unsigned short* __restrict__ kvWbf,  // [768][1152]
                           const float* __restrict__ kv_b,
                           const float* __restrict__ qkw,             // [8][768]
                           const float* __restrict__ qb,              // [8]
                           unsigned short* __restrict__ KVt,          // [B][64][768][32]
                           float* __restrict__ scores)                // [B][H][N]
{
  __shared__ unsigned short sA[2][32 * 32];   // [row][k]  2 x 2 KB
  __shared__ unsigned short sB[2][Dx * 32];   // [col][k]  2 x 48 KB (TDM dst)
  __shared__ float sScore[32 * Hx];           // 1 KB

  const int b     = blockIdx.y;
  const int chunk = blockIdx.x;            // 64 chunks of 32 rows
  const int nBase = chunk * 32;
  const int tid   = threadIdx.x;
  const int wave  = tid >> 5;
  const int lane  = tid & 31;
  const int mt    = wave & 1;              // m-tile within the 32 rows
  const int ng    = wave >> 1;             // column group (0..3) * 192

  if (tid < 32 * Hx) sScore[tid] = 0.f;

  const float w0 = t2v_w0[0], b0 = t2v_b0[0];
  const unsigned sB_lds0 = (unsigned)(unsigned long long)&sB[0][0];
  const unsigned sB_lds1 = (unsigned)(unsigned long long)&sB[1][0];

  // Stage one A-tile (32 rows x 32 k) for chunk kcs into dstbuf:
  // text_embs for k<768, time2vec features (lin / sin) beyond.
  auto stage_a = [&](int kcs, unsigned short* dstbuf) {
    const int k0s = kcs * 32;
    const int r   = tid >> 3;          // row 0..31
    const int kk  = (tid & 7) * 4;     // 4 elems per thread
    const int n   = nBase + r;
    unsigned short* dst = dstbuf + r * 32 + kk;
    if (k0s < Dx) {
      const float4 f4 = *reinterpret_cast<const float4*>(
          text_embs + ((size_t)b * Nx + n) * Dx + k0s + kk);
      dst[0] = f2bf(f4.x); dst[1] = f2bf(f4.y);
      dst[2] = f2bf(f4.z); dst[3] = f2bf(f4.w);
      // prefetch two chunks ahead (one ahead is staged by double buffering)
      if (k0s + 64 < Dx)
        __builtin_prefetch(text_embs + ((size_t)b * Nx + n) * Dx + k0s + 64 + kk, 0, 0);
    } else {
      const float t = text_ts[(size_t)b * Nx + n];
#pragma unroll
      for (int q = 0; q < 4; ++q) {
        const int j = k0s + kk + q - Dx;           // 0..383
        const float vf = (j == 0) ? (t * w0 + b0)
                                  : __sinf(t * t2v_Wp[j - 1] + t2v_bp[j - 1]);
        dst[q] = f2bf(vf);
      }
    }
  };

  FAcc acc[12];
#pragma unroll
  for (int i = 0; i < 12; ++i) acc[i].v = zero8();

  // Prologue: chunk 0 into buffer 0.
  if (wave == 0)
    tdm_load_2d(sB_lds0, kvWbf, /*2B*/ 1, /*tile0*/ 32, /*tile1*/ Dx, /*stride*/ Fx);
  stage_a(0, &sA[0][0]);

  for (int kc = 0; kc < 36; ++kc) {
    const int cur = kc & 1;
    if (wave == 0) __builtin_amdgcn_s_wait_tensorcnt(0);   // sB[cur] DMA done
    __syncthreads();                                       // publish sA/sB[cur]

    // Kick off next chunk into the other buffer while we compute.
    if (kc + 1 < 36) {
      if (wave == 0)
        tdm_load_2d(cur ? sB_lds0 : sB_lds1, kvWbf + (size_t)(kc + 1) * 32,
                    1, 32, Dx, Fx);
      stage_a(kc + 1, &sA[cur ^ 1][0]);
    }

    // A fragment: 16x32 bf16. lane L: row = m0+(L&15); K-half = L>>4.
    BFrag a;
    {
      const int row   = mt * 16 + (lane & 15);
      const int khalf = lane >> 4;
      const unsigned short* base = &sA[cur][0] + row * 32;
#pragma unroll
      for (int i = 0; i < 8; ++i) {
        const int k = khalf * 16 + i * 2;
        a.u[i] = *reinterpret_cast<const unsigned int*>(base + k);
      }
    }

    // Load all B fragments, then run the WMMAs back-to-back.
    BFrag bf[12];
#pragma unroll
    for (int tno = 0; tno < 12; ++tno) {
      const int col = ng * 192 + tno * 16 + (lane & 15);
      const uint4* bp4 =
          reinterpret_cast<const uint4*>(&sB[cur][0] + col * 32 + (lane >> 4) * 16);
      const uint4 x0 = bp4[0], x1 = bp4[1];
      bf[tno].u[0] = x0.x; bf[tno].u[1] = x0.y; bf[tno].u[2] = x0.z; bf[tno].u[3] = x0.w;
      bf[tno].u[4] = x1.x; bf[tno].u[5] = x1.y; bf[tno].u[6] = x1.z; bf[tno].u[7] = x1.w;
    }
#pragma unroll
    for (int tno = 0; tno < 12; ++tno)
      acc[tno].v = wmma_bf16(a, bf[tno], acc[tno].v);
  }

  // Epilogue: + kv_b, store KV bf16 (chunk-transposed, non-temporal),
  // accumulate score partials into LDS with ds_add_f32.
  float part[8][8];
#pragma unroll
  for (int r = 0; r < 8; ++r)
#pragma unroll
    for (int h = 0; h < Hx; ++h) part[r][h] = 0.f;

#pragma unroll
  for (int tno = 0; tno < 12; ++tno) {
    const int col = ng * 192 + tno * 16 + (lane & 15);
    const float bias = kv_b[col];
    float qw[8];
#pragma unroll
    for (int h = 0; h < Hx; ++h) qw[h] = qkw[h * Dx + col];
#pragma unroll
    for (int r = 0; r < 8; ++r) {
      const int m = mt * 16 + r + (lane >> 4) * 8;
      const float val = acc[tno].f[r] + bias;
      __builtin_nontemporal_store(
          f2bf(val), &KVt[(((size_t)b * 64 + chunk) * Dx + col) * 32 + m]);
#pragma unroll
      for (int h = 0; h < Hx; ++h) part[r][h] += val * qw[h];
    }
  }
#pragma unroll
  for (int r = 0; r < 8; ++r) {
    const int m = mt * 16 + r + (lane >> 4) * 8;
#pragma unroll
    for (int h = 0; h < Hx; ++h) atomicAdd(&sScore[m * Hx + h], part[r][h]);
  }
  __syncthreads();
  {
    const float scale = 0.1020620726159658f;   // 1/sqrt(96)
    const int m = tid >> 3, h = tid & 7;       // 256 = 32*8 exactly
    scores[((size_t)b * Hx + h) * Nx + nBase + m] = scale * (sScore[tid] + qb[h]);
  }
}

// ---------------------------------------------------------------------------
// Kernel 2: masked softmax over N per (b,h); attn stored bf16.
// ---------------------------------------------------------------------------
__global__ __launch_bounds__(256)
void softmax_kernel(const float* __restrict__ scores,
                    const unsigned char* __restrict__ mask,
                    unsigned short* __restrict__ attnbf) {
  __shared__ float red[256];
  const int bh = blockIdx.x;         // b*H + h
  const int b  = bh >> 3;
  const int tid = threadIdx.x;
  const float* row = scores + (size_t)bh * Nx;
  const unsigned char* mrow = mask + (size_t)b * Nx;

  float mx = -1e30f;
  for (int n = tid; n < Nx; n += 256)
    if (mrow[n]) mx = fmaxf(mx, row[n]);
  red[tid] = mx; __syncthreads();
  for (int s = 128; s > 0; s >>= 1) {
    if (tid < s) red[tid] = fmaxf(red[tid], red[tid + s]);
    __syncthreads();
  }
  mx = red[0]; __syncthreads();

  float sum = 0.f;
  for (int n = tid; n < Nx; n += 256)
    if (mrow[n]) sum += __expf(row[n] - mx);
  red[tid] = sum; __syncthreads();
  for (int s = 128; s > 0; s >>= 1) {
    if (tid < s) red[tid] += red[tid + s];
    __syncthreads();
  }
  const float inv = 1.f / red[0];    // mask[:,0] == True guarantees > 0

  for (int n = tid; n < Nx; n += 256) {
    const float a = mrow[n] ? __expf(row[n] - mx) * inv : 0.f;
    attnbf[(size_t)bh * Nx + n] = f2bf(a);
  }
}

// ---------------------------------------------------------------------------
// Kernel 3: ctxKV[b,h,:] = attn[b,h,:] @ KV[b,:,:] via WMMA bf16.
// KV chunks (48 KB contiguous in the transposed layout) double-buffered via
// TDM. One block per b; wave w owns d-tiles w*6 .. w*6+5; 64 K-chunks.
// ---------------------------------------------------------------------------
__global__ __launch_bounds__(256)
void ctx_gemm_kernel(const unsigned short* __restrict__ attnbf,
                     const unsigned short* __restrict__ KVt,   // [B][64][768][32]
                     float* __restrict__ ctxKV)                // [B][8][768]
{
  __shared__ unsigned short sKV[2][Dx * 32];   // [d][n] 2 x 48 KB (TDM dst)
  const int b = blockIdx.x;
  const int tid = threadIdx.x;
  const int wave = tid >> 5, lane = tid & 31;
  const unsigned lds0 = (unsigned)(unsigned long long)&sKV[0][0];
  const unsigned lds1 = (unsigned)(unsigned long long)&sKV[1][0];

  FAcc acc[6];
#pragma unroll
  for (int i = 0; i < 6; ++i) acc[i].v = zero8();

  // Prologue: chunk 0 into buffer 0 (49152B = flat tile of 6144 x 8B).
  if (wave == 0)
    tdm_load_2d(lds0, KVt + ((size_t)b * 64 * Dx) * 32, /*8B*/ 3,
                /*tile0*/ 6144, /*tile1*/ 1, /*stride*/ 6144);

  for (int kc = 0; kc < 64; ++kc) {
    const int cur = kc & 1;
    if (wave == 0) __builtin_amdgcn_s_wait_tensorcnt(0);
    __syncthreads();
    if (kc + 1 < 64 && wave == 0)
      tdm_load_2d(cur ? lds0 : lds1,
                  KVt + (((size_t)b * 64 + kc + 1) * Dx) * 32,
                  3, 6144, 1, 6144);

    // A fragment: row = head (lanes with row>=8 are zero padding)
    BFrag a;
    {
      const int hrow = lane & 15, khalf = lane >> 4;
      if (hrow < Hx) {
        const unsigned short* abase =
            attnbf + ((size_t)b * Hx + hrow) * Nx + kc * 32;
#pragma unroll
        for (int i = 0; i < 8; ++i) {
          const int k = khalf * 16 + i * 2;
          a.u[i] = *reinterpret_cast<const unsigned int*>(abase + k);
        }
      } else {
#pragma unroll
        for (int i = 0; i < 8; ++i) a.u[i] = 0u;
      }
    }

    BFrag bf[6];
#pragma unroll
    for (int t = 0; t < 6; ++t) {
      const int d = (wave * 6 + t) * 16 + (lane & 15);
      const uint4* bp4 =
          reinterpret_cast<const uint4*>(&sKV[cur][0] + d * 32 + (lane >> 4) * 16);
      const uint4 x0 = bp4[0], x1 = bp4[1];
      bf[t].u[0] = x0.x; bf[t].u[1] = x0.y; bf[t].u[2] = x0.z; bf[t].u[3] = x0.w;
      bf[t].u[4] = x1.x; bf[t].u[5] = x1.y; bf[t].u[6] = x1.z; bf[t].u[7] = x1.w;
    }
#pragma unroll
    for (int t = 0; t < 6; ++t)
      acc[t].v = wmma_bf16(a, bf[t], acc[t].v);
  }
  // D layout: VGPR r, lane L -> row = r + (L>>4)*8, col = (L&15). Keep rows<8.
  if (lane < 16) {
#pragma unroll
    for (int t = 0; t < 6; ++t) {
      const int d = (wave * 6 + t) * 16 + lane;
#pragma unroll
      for (int r = 0; r < 8; ++r)
        ctxKV[((size_t)b * Hx + r) * Dx + d] = acc[t].f[r];
    }
  }
}

// ---------------------------------------------------------------------------
// Kernel 4: Wv contraction + out_proj + residual + LayerNorm + po + gate.
// ---------------------------------------------------------------------------
__global__ __launch_bounds__(256)
void epilogue_kernel(const float* __restrict__ ctxKV,
                     const float* __restrict__ in_w, const float* __restrict__ in_b,
                     const float* __restrict__ out_w, const float* __restrict__ out_b,
                     const float* __restrict__ Qp,
                     const float* __restrict__ ln_g, const float* __restrict__ ln_b,
                     const float* __restrict__ po_W, const float* __restrict__ po_b,
                     const unsigned char* __restrict__ mask,
                     float* __restrict__ out) {
  __shared__ float sc[Dx];
  __shared__ float sx[Dx];
  __shared__ float red[256];
  __shared__ int sAny;
  const int b = blockIdx.x, tid = threadIdx.x;

  if (tid == 0) sAny = 0;
  __syncthreads();
  {
    int any = 0;
    for (int n = tid; n < Nx; n += 256) any |= mask[(size_t)b * Nx + n];
    if (any) atomicOr(&sAny, 1);
  }

  // ctx[d] = ctxKV[b, d/96, :] . Wv[d,:] + bv[d]
  for (int d = tid; d < Dx; d += 256) {
    const int h = d / HDx;
    const float* c = ctxKV + ((size_t)b * Hx + h) * Dx;
    const float* w = in_w + (size_t)(2 * Dx + d) * Dx;
    float a = in_b[2 * Dx + d];
    for (int f = 0; f < Dx; ++f) a += c[f] * w[f];
    sc[d] = a;
  }
  __syncthreads();

  // x = out_proj(ctx) + Qp
  for (int e = tid; e < Dx; e += 256) {
    const float* w = out_w + (size_t)e * Dx;
    float a = out_b[e];
    for (int d = 0; d < Dx; ++d) a += sc[d] * w[d];
    sx[e] = a + Qp[e];
  }
  __syncthreads();

  // LayerNorm
  float s = 0.f;
  for (int e = tid; e < Dx; e += 256) s += sx[e];
  red[tid] = s; __syncthreads();
  for (int w = 128; w > 0; w >>= 1) {
    if (tid < w) red[tid] += red[tid + w];
    __syncthreads();
  }
  const float mu = red[0] / (float)Dx;
  __syncthreads();
  float s2 = 0.f;
  for (int e = tid; e < Dx; e += 256) { const float d = sx[e] - mu; s2 += d * d; }
  red[tid] = s2; __syncthreads();
  for (int w = 128; w > 0; w >>= 1) {
    if (tid < w) red[tid] += red[tid + w];
    __syncthreads();
  }
  const float inv = rsqrtf(red[0] / (float)Dx + 1e-5f);
  __syncthreads();
  for (int e = tid; e < Dx; e += 256)
    sc[e] = (sx[e] - mu) * inv * ln_g[e] + ln_b[e];
  __syncthreads();

  const float gate = sAny ? 1.f : 0.f;
  for (int e2 = tid; e2 < Dx; e2 += 256) {
    const float* w = po_W + (size_t)e2 * Dx;
    float a = po_b[e2];
    for (int e = 0; e < Dx; ++e) a += sc[e] * w[e];
    out[(size_t)b * Dx + e2] = a * gate;
  }
}

// ---------------------------------------------------------------------------
extern "C" void kernel_launch(void* const* d_in, const int* in_sizes, int n_in,
                              void* d_out, int out_size, void* d_ws, size_t ws_size,
                              hipStream_t stream) {
  (void)in_sizes; (void)n_in; (void)out_size; (void)ws_size;
  const float* text_embs = (const float*)d_in[0];
  const float* text_ts   = (const float*)d_in[1];
  const unsigned char* text_mask = (const unsigned char*)d_in[3];
  const float* t2v_w0 = (const float*)d_in[4];
  const float* t2v_b0 = (const float*)d_in[5];
  const float* t2v_Wp = (const float*)d_in[6];
  const float* t2v_bp = (const float*)d_in[7];
  const float* kv_W   = (const float*)d_in[8];
  const float* kv_b   = (const float*)d_in[9];
  const float* Qp     = (const float*)d_in[10];
  const float* in_w   = (const float*)d_in[11];
  const float* in_b   = (const float*)d_in[12];
  const float* out_w  = (const float*)d_in[13];
  const float* out_b  = (const float*)d_in[14];
  const float* ln_g   = (const float*)d_in[15];
  const float* ln_b   = (const float*)d_in[16];
  const float* po_W   = (const float*)d_in[17];
  const float* po_b   = (const float*)d_in[18];
  float* out = (float*)d_out;

  char* ws = (char*)d_ws;
  constexpr size_t OFF_KVW = 0;
  constexpr size_t SZ_KVW  = (size_t)Dx * Fx * 2;                 // kv_W bf16
  constexpr size_t OFF_Q   = OFF_KVW + SZ_KVW;                    // q f32 [768]
  constexpr size_t OFF_QKW = OFF_Q + (size_t)Dx * 4;              // qk_w [8][768]
  constexpr size_t OFF_QB  = OFF_QKW + (size_t)Hx * Dx * 4;       // qb [8]
  constexpr size_t OFF_KVT = ((OFF_QB + Hx * 4 + 255) / 256) * 256;
  constexpr size_t SZ_KVT  = (size_t)Bx * Nx * Dx * 2;            // KV bf16
  constexpr size_t OFF_SC  = OFF_KVT + SZ_KVT;                    // scores f32
  constexpr size_t SZ_SC   = (size_t)Bx * Hx * Nx * 4;
  constexpr size_t OFF_ATT = OFF_SC + SZ_SC;                      // attn bf16
  constexpr size_t SZ_ATT  = (size_t)Bx * Hx * Nx * 2;
  constexpr size_t OFF_CTX = OFF_ATT + SZ_ATT;                    // ctxKV f32

  unsigned short* kvWbf = (unsigned short*)(ws + OFF_KVW);
  float* qv   = (float*)(ws + OFF_Q);
  float* qkw  = (float*)(ws + OFF_QKW);
  float* qb   = (float*)(ws + OFF_QB);
  unsigned short* KVt   = (unsigned short*)(ws + OFF_KVT);
  float* scores = (float*)(ws + OFF_SC);
  unsigned short* attnbf = (unsigned short*)(ws + OFF_ATT);
  float* ctxKV = (float*)(ws + OFF_CTX);

  prep_q_kernel<<<1, 256, 0, stream>>>(Qp, in_w, in_b, qv, qkw, qb);
  cvt_bf16_kernel<<<(Dx * Fx + 255) / 256, 256, 0, stream>>>(kv_W, kvWbf, Dx * Fx);
  kv_gemm_scores_kernel<<<dim3(Nx / 32, Bx), 256, 0, stream>>>(
      text_embs, text_ts, t2v_w0, t2v_b0, t2v_Wp, t2v_bp,
      kvWbf, kv_b, qkw, qb, KVt, scores);
  softmax_kernel<<<Bx * Hx, 256, 0, stream>>>(scores, text_mask, attnbf);
  ctx_gemm_kernel<<<Bx, 256, 0, stream>>>(attnbf, KVt, ctxKV);
  epilogue_kernel<<<Bx, 256, 0, stream>>>(ctxKV, in_w, in_b, out_w, out_b, Qp,
                                          ln_g, ln_b, po_W, po_b, text_mask, out);
}
